// TransformerModel_40827959116576
// MI455X (gfx1250) — matrix operand order repaired and checked by reading, not compiled, not run on hardware
//
#include <hip/hip_runtime.h>
#include <hip/hip_bf16.h>
#include <math.h>

// ---------------------------------------------------------------------------
// CDNA5 (gfx1250) transformer forward.  bf16 WMMA (16x16x32) with f32 accum.
// Software-pipelined global->reg->LDS staging (cvt f32->bf16 on the way in).
// ---------------------------------------------------------------------------

typedef __attribute__((ext_vector_type(16))) __bf16 v16bf;
typedef __attribute__((ext_vector_type(8)))  __bf16 v8bf;
typedef __attribute__((ext_vector_type(8)))  float  v8f;

// ---------------------------------------------------------------------------
// Generic tiled GEMM:  C[M,N] = act( A[M,K] * op(B) + bias )
//   - A row-major, lda;  B row-major [K,N] (TRB=0) or [N,K] (TRB=1)
//   - f32 in global, converted to bf16 while staging through LDS
//   - batched via blockIdx.z with (outer,inner) strides: z = b*innerH + h
//   - 256 threads = 8 wave32s, arranged WR x WC; wave tile (BM/WR) x (BN/WC)
//   - register double-buffer: next k-tile's global loads issue during WMMA
// ---------------------------------------------------------------------------
template<int BM, int BN, int WR, int WC, int TRB>
__global__ __launch_bounds__(256)
void gemm_wmma_kernel(const float* __restrict__ A, const float* __restrict__ Bm,
                      float* __restrict__ C, const float* __restrict__ bias,
                      int K, int lda, int ldb, int ldc,
                      long sAb, long sAh, long sBb, long sBh, long sCb, long sCh,
                      int innerH, int relu)
{
  constexpr int TM  = BM / WR;         // wave tile rows
  constexpr int TN  = BN / WC;         // wave tile cols
  constexpr int MI  = TM / 16;         // 16x16 WMMA tiles per wave (rows)
  constexpr int NI  = TN / 16;         // 16x16 WMMA tiles per wave (cols)
  constexpr int LP  = 40;              // bf16 LDS row pitch: 80B, 16B-aligned
  constexpr int ANL = BM / 32;         // float4 loads per thread for A tile
  constexpr int BNL = BN / 32;         // float4 loads per thread for B tile

  __shared__ __align__(16) __bf16 lsA[BM][LP];   // A tile [BM][32] bf16
  __shared__ __align__(16) __bf16 lsB[BN][LP];   // B tile [BN][32] bf16 ([n][k])

  {
    int z  = blockIdx.z;
    int bb = z / innerH;
    int hh = z - bb * innerH;
    A  += bb * sAb + hh * sAh;
    Bm += bb * sBb + hh * sBh;
    C  += bb * sCb + hh * sCh;
  }

  const int n0 = blockIdx.x * BN;
  const int m0 = blockIdx.y * BM;

  const int t    = threadIdx.x;
  const int wave = t >> 5;             // wave32
  const int lane = t & 31;
  const int wr   = wave / WC;
  const int wc   = wave - wr * WC;
  const int wm   = wr * TM;
  const int wn   = wc * TN;
  const int col  = lane & 15;
  const int hi16 = lane >> 4;          // lanes 16..31 -> second K/M half

  // ---- precompute per-thread staging pointers (advance by fixed stride) ----
  const float* ap[ANL];
  __bf16*      asd[ANL];
#pragma unroll
  for (int i = 0; i < ANL; ++i) {
    int idx = t + i * 256;
    int row = idx >> 3;
    int c   = (idx & 7) << 2;
    ap[i]  = A + (long)(m0 + row) * lda + c;   // k advances by +=32
    asd[i] = &lsA[row][c];
  }
  const float* bp[BNL];
  __bf16*      bsd0[BNL];              // TRB=1: row dest; TRB=0: first of 4 col dests
  int          bkk[BNL];               // TRB=0: k index within tile
#pragma unroll
  for (int i = 0; i < BNL; ++i) {
    int idx = t + i * 256;
    if (TRB) {                         // B is [N,K]: copy BN rows x 32
      int row = idx >> 3;
      int c   = (idx & 7) << 2;
      bp[i]   = Bm + (long)(n0 + row) * ldb + c;      // += 32 per step
      bsd0[i] = &lsB[row][c];
      bkk[i]  = 0;
    } else {                           // B is [K,N]: transpose 32 x BN block
      int kk = idx / (BN / 4);
      int c  = (idx - kk * (BN / 4)) << 2;
      bp[i]   = Bm + (long)kk * ldb + n0 + c;         // += 32*ldb per step
      bsd0[i] = &lsB[c][kk];
      bkk[i]  = kk;
    }
  }

  v8f acc[MI][NI];
#pragma unroll
  for (int mi = 0; mi < MI; ++mi)
#pragma unroll
    for (int ni = 0; ni < NI; ++ni)
      acc[mi][ni] = (v8f){};

  float4 ra[ANL], rb[BNL];

  // issue all global loads for one k-tile into registers (no waits between)
  auto issue_loads = [&]() {
#pragma unroll
    for (int i = 0; i < ANL; ++i) { ra[i] = *(const float4*)ap[i]; ap[i] += 32; }
#pragma unroll
    for (int i = 0; i < BNL; ++i) {
      rb[i] = *(const float4*)bp[i];
      bp[i] += TRB ? 32 : (long)32 * ldb;
    }
  };
  // convert staged registers to bf16 and store into LDS
  auto store_tiles = [&]() {
#pragma unroll
    for (int i = 0; i < ANL; ++i) {
      asd[i][0] = (__bf16)ra[i].x;  asd[i][1] = (__bf16)ra[i].y;
      asd[i][2] = (__bf16)ra[i].z;  asd[i][3] = (__bf16)ra[i].w;
    }
#pragma unroll
    for (int i = 0; i < BNL; ++i) {
      if (TRB) {
        bsd0[i][0] = (__bf16)rb[i].x;  bsd0[i][1] = (__bf16)rb[i].y;
        bsd0[i][2] = (__bf16)rb[i].z;  bsd0[i][3] = (__bf16)rb[i].w;
      } else {                         // scatter 4 columns, same k
        bsd0[i][0 * LP] = (__bf16)rb[i].x;
        bsd0[i][1 * LP] = (__bf16)rb[i].y;
        bsd0[i][2 * LP] = (__bf16)rb[i].z;
        bsd0[i][3 * LP] = (__bf16)rb[i].w;
      }
    }
  };

  const int nsteps = K >> 5;           // K / 32
  issue_loads();                       // prologue: k-tile 0 in flight

  for (int s = 0; s < nsteps; ++s) {
    store_tiles();                     // consumes in-flight loads (one wait)
    __syncthreads();                   // tiles visible to all waves
    if (s + 1 < nsteps) issue_loads(); // next k-tile overlaps WMMA below

    // ---- build fragments (ISA 7.12.2 bf16 A 16x32 / B 32x16 layouts)
    v16bf af[MI];
    const int aklo = hi16 ? 8 : 0;     // lanes16-31 hold K=8..15 / 24..31
#pragma unroll
    for (int mi = 0; mi < MI; ++mi) {
      int row = wm + mi * 16 + col;    // A: lane%16 = M row
      v8bf lo = *(const v8bf*)&lsA[row][aklo];       // K = aklo..aklo+7
      v8bf hi = *(const v8bf*)&lsA[row][16 + aklo];  // K = 16+aklo..23+aklo
#pragma unroll
      for (int j = 0; j < 8; ++j) { af[mi][j] = lo[j]; af[mi][8 + j] = hi[j]; }
    }
    v16bf bfr[NI];
    const int bk = hi16 ? 8 : 0;       // lanes16-31 hold K=8..15 / 24..31
#pragma unroll
    for (int ni = 0; ni < NI; ++ni) {
      int bc = wn + ni * 16 + col;     // B: lane%16 = N col
      v8bf l0 = *(const v8bf*)&lsB[bc][bk];
      v8bf l1 = *(const v8bf*)&lsB[bc][bk + 16];
#pragma unroll
      for (int j = 0; j < 8; ++j) { bfr[ni][j] = l0[j]; bfr[ni][8 + j] = l1[j]; }
    }

#pragma unroll
    for (int mi = 0; mi < MI; ++mi)
#pragma unroll
      for (int ni = 0; ni < NI; ++ni)
        acc[mi][ni] = __builtin_amdgcn_wmma_f32_16x16x32_bf16(
            false, af[mi], false, bfr[ni], (short)0, acc[mi][ni], false, false);

    __syncthreads();                   // all reads done before next store
  }

  // ---- epilogue: C/D layout — VGPR r -> M = r + 8*hi16, lane%16 -> N
  const int rbase = hi16 * 8;
#pragma unroll
  for (int mi = 0; mi < MI; ++mi)
#pragma unroll
    for (int ni = 0; ni < NI; ++ni) {
      int gn = n0 + wn + ni * 16 + col;
      float bv = bias ? bias[gn] : 0.0f;
#pragma unroll
      for (int r = 0; r < 8; ++r) {
        int gm = m0 + wm + mi * 16 + rbase + r;
        float vv = acc[mi][ni][r] + bv;
        if (relu) vv = fmaxf(vv, 0.0f);
        C[(long)gm * ldc + gn] = vv;
      }
    }
}

// ---------------------------------------------------------------------------
// Embedding gather * sqrt(D) + sinusoidal positional encoding
// ---------------------------------------------------------------------------
__global__ __launch_bounds__(256)
void embed_kernel(const int* __restrict__ tok, const float* __restrict__ emb,
                  float* __restrict__ out, int L, int Dm, float scale)
{
  int row = blockIdx.x;                // b*L + pos
  int pos = row % L;
  int token = tok[row];
  const float* e = emb + (long)token * Dm;
  float* o = out + (long)row * Dm;
  for (int d = threadIdx.x; d < Dm; d += blockDim.x) {
    int i = d >> 1;
    float ang = (float)pos * __expf(-(2.0f * (float)i / (float)Dm) * 9.210340372f);
    float pe = (d & 1) ? __cosf(ang) : __sinf(ang);
    o[d] = e[d] * scale + pe;
  }
}

// ---------------------------------------------------------------------------
// Masked + scaled softmax over one score row (pad mask on keys, opt. causal).
// Matches ref: scores = where(mask,-inf,scores); softmax(scores * scale)
// ---------------------------------------------------------------------------
__global__ __launch_bounds__(256)
void softmax_mask_kernel(float* __restrict__ scores, const int* __restrict__ keys,
                         int Hn, int Lq, int Lk, int causal, float scale)
{
  int q = blockIdx.x;
  int z = blockIdx.y;                  // b*H + h
  int b = z / Hn;
  float* row = scores + ((long)z * Lq + q) * Lk;
  const int* krow = keys + (long)b * Lk;

  __shared__ float red[256];
  int t = threadIdx.x;
  int per = Lk / 256;                  // 2 here
  float vals[4];
  float vmax = -INFINITY;
  for (int i = 0; i < per; ++i) {
    int j = t + i * 256;
    bool m = (krow[j] == 0) || (causal && j > q);
    float v = m ? -INFINITY : row[j] * scale;
    vals[i] = v;
    vmax = fmaxf(vmax, v);
  }
  red[t] = vmax; __syncthreads();
  for (int o = 128; o > 0; o >>= 1) { if (t < o) red[t] = fmaxf(red[t], red[t + o]); __syncthreads(); }
  vmax = red[0]; __syncthreads();

  float sum = 0.0f;
  for (int i = 0; i < per; ++i) { vals[i] = __expf(vals[i] - vmax); sum += vals[i]; }
  red[t] = sum; __syncthreads();
  for (int o = 128; o > 0; o >>= 1) { if (t < o) red[t] += red[t + o]; __syncthreads(); }
  float inv = 1.0f / red[0];
  for (int i = 0; i < per; ++i) row[t + i * 256] = vals[i] * inv;
}

// ---------------------------------------------------------------------------
// Fused residual add + LayerNorm (in-place on x):  x = LN(x + y) * g + b
// ---------------------------------------------------------------------------
__global__ __launch_bounds__(256)
void add_ln_kernel(float* __restrict__ x, const float* __restrict__ y,
                   const float* __restrict__ g, const float* __restrict__ b, int Dm)
{
  long row = blockIdx.x;
  float* xr = x + row * Dm;
  const float* yr = y + row * Dm;
  __shared__ float red[256];
  int t = threadIdx.x;
  int per = Dm / 256;                  // 2 here
  float s[4];
  float local = 0.0f;
  for (int i = 0; i < per; ++i) { s[i] = xr[t + i * 256] + yr[t + i * 256]; local += s[i]; }
  red[t] = local; __syncthreads();
  for (int o = 128; o > 0; o >>= 1) { if (t < o) red[t] += red[t + o]; __syncthreads(); }
  float mean = red[0] / (float)Dm; __syncthreads();
  local = 0.0f;
  for (int i = 0; i < per; ++i) { float d = s[i] - mean; local += d * d; }
  red[t] = local; __syncthreads();
  for (int o = 128; o > 0; o >>= 1) { if (t < o) red[t] += red[t + o]; __syncthreads(); }
  float inv = rsqrtf(red[0] / (float)Dm + 1e-5f);
  for (int i = 0; i < per; ++i) {
    int j = t + i * 256;
    xr[j] = (s[i] - mean) * inv * g[j] + b[j];
  }
}

// ---------------------------------------------------------------------------
// Host-side GEMM dispatch (tile config by N, transB as template param)
// ---------------------------------------------------------------------------
static inline void gemm(hipStream_t s,
                        const float* A, const float* B, float* C, const float* bias,
                        int M, int N, int K, int lda, int ldb, int ldc,
                        long sAb, long sAh, long sBb, long sBh, long sCb, long sCh,
                        int innerH, int batches, int transB, int relu)
{
  if ((N & 127) == 0) {
    dim3 g(N / 128, M / 128, batches);
    if (transB)
      gemm_wmma_kernel<128, 128, 2, 4, 1><<<g, 256, 0, s>>>(
          A, B, C, bias, K, lda, ldb, ldc, sAb, sAh, sBb, sBh, sCb, sCh, innerH, relu);
    else
      gemm_wmma_kernel<128, 128, 2, 4, 0><<<g, 256, 0, s>>>(
          A, B, C, bias, K, lda, ldb, ldc, sAb, sAh, sBb, sBh, sCb, sCh, innerH, relu);
  } else {          // N multiple of 64 (attention P*V, N = DK = 64); always NN
    dim3 g(N / 64, M / 128, batches);
    gemm_wmma_kernel<128, 64, 4, 2, 0><<<g, 256, 0, s>>>(
        A, B, C, bias, K, lda, ldb, ldc, sAb, sAh, sBb, sBh, sCb, sCh, innerH, relu);
  }
}

// ---------------------------------------------------------------------------
extern "C" void kernel_launch(void* const* d_in, const int* in_sizes, int n_in,
                              void* d_out, int out_size, void* d_ws, size_t ws_size,
                              hipStream_t stream)
{
  (void)in_sizes; (void)n_in; (void)out_size; (void)ws_size;

  constexpr int Bc = 4, Sc = 512, Tc = 512, Dc = 512, Fc = 2048;
  constexpr int Hc = 8, DKc = 64, Nl = 6, HDK = Hc * DKc;       // 512
  constexpr int Vc = 32000;
  const int MS = Bc * Sc;                                        // 2048 rows

  // ---- inputs (setup_inputs() order) ----
  const int*   src      = (const int*)d_in[0];
  const int*   trg      = (const int*)d_in[1];
  const float* emb_src  = (const float*)d_in[2];
  const float* emb_trg  = (const float*)d_in[3];
  const float* enc_wq   = (const float*)d_in[4];
  const float* enc_wk   = (const float*)d_in[5];
  const float* enc_wv   = (const float*)d_in[6];
  const float* enc_wo   = (const float*)d_in[7];
  const float* enc_ln1g = (const float*)d_in[8];
  const float* enc_ln1b = (const float*)d_in[9];
  const float* enc_w1   = (const float*)d_in[10];
  const float* enc_b1   = (const float*)d_in[11];
  const float* enc_w2   = (const float*)d_in[12];
  const float* enc_b2   = (const float*)d_in[13];
  const float* enc_ln2g = (const float*)d_in[14];
  const float* enc_ln2b = (const float*)d_in[15];
  const float* dec_wq   = (const float*)d_in[16];
  const float* dec_wk   = (const float*)d_in[17];
  const float* dec_wv   = (const float*)d_in[18];
  const float* dec_wo   = (const float*)d_in[19];
  const float* dec_ln1g = (const float*)d_in[20];
  const float* dec_ln1b = (const float*)d_in[21];
  const float* dec_cq   = (const float*)d_in[22];
  const float* dec_ck   = (const float*)d_in[23];
  const float* dec_cv   = (const float*)d_in[24];
  const float* dec_co   = (const float*)d_in[25];
  const float* dec_ln2g = (const float*)d_in[26];
  const float* dec_ln2b = (const float*)d_in[27];
  const float* dec_w1   = (const float*)d_in[28];
  const float* dec_b1   = (const float*)d_in[29];
  const float* dec_w2   = (const float*)d_in[30];
  const float* dec_b2   = (const float*)d_in[31];
  const float* dec_ln3g = (const float*)d_in[32];
  const float* dec_ln3b = (const float*)d_in[33];
  const float* proj     = (const float*)d_in[34];
  float* out = (float*)d_out;

  // ---- workspace layout (floats) ----
  float* ws = (float*)d_ws;
  const long ACT = (long)MS * Dc;                 // 1,048,576
  float* X  = ws;                                 // encoder activations
  float* Y  = X  + ACT;                           // decoder activations
  float* T0 = Y  + ACT;                           // matmul temp
  float* Q  = T0 + ACT;
  float* Kb = Q  + ACT;
  float* Vb = Kb + ACT;
  float* AT = Vb + ACT;                           // attention output (concat heads)
  float* FF = AT + ACT;                           // FFN hidden [MS, F]
  float* SC = FF + (long)MS * Fc;                 // scores [B*H, 512, 512]

  const long sTok = (long)Sc * HDK;               // per-batch stride in q/k/v
  const long sScB = (long)Hc * Sc * Sc;           // per-batch stride in scores
  const long sScH = (long)Sc * Sc;                // per-head stride in scores
  const float att_scale = 0.125f;                 // DK^-0.5
  const float emb_scale = 22.627416998f;          // sqrt(D)

  // =============================== encoder ===============================
  embed_kernel<<<MS, 256, 0, stream>>>(src, emb_src, X, Sc, Dc, emb_scale);

  for (int l = 0; l < Nl; ++l) {
    const float* wq = enc_wq + (long)l * Dc * HDK;
    const float* wk = enc_wk + (long)l * Dc * HDK;
    const float* wv = enc_wv + (long)l * Dc * HDK;
    const float* wo = enc_wo + (long)l * HDK * Dc;
    const float* w1 = enc_w1 + (long)l * Dc * Fc;
    const float* w2 = enc_w2 + (long)l * Fc * Dc;

    gemm(stream, X, wq, Q,  nullptr, MS, HDK, Dc, Dc, HDK, HDK, 0,0,0,0,0,0, 1, 1, 0, 0);
    gemm(stream, X, wk, Kb, nullptr, MS, HDK, Dc, Dc, HDK, HDK, 0,0,0,0,0,0, 1, 1, 0, 0);
    gemm(stream, X, wv, Vb, nullptr, MS, HDK, Dc, Dc, HDK, HDK, 0,0,0,0,0,0, 1, 1, 0, 0);

    // scores[b,h] = Q[b,h] (512x64) * K[b,h]^T  -> [512,512]
    gemm(stream, Q, Kb, SC, nullptr, Sc, Sc, DKc, HDK, HDK, Sc,
         sTok, DKc, sTok, DKc, sScB, sScH, Hc, Bc * Hc, 1, 0);
    softmax_mask_kernel<<<dim3(Sc, Bc * Hc), 256, 0, stream>>>(
        SC, src, Hc, Sc, Sc, 0, att_scale);
    // attn[b,h] = P (512x512) * V[b,h] (512x64)
    gemm(stream, SC, Vb, AT, nullptr, Sc, DKc, Sc, Sc, HDK, HDK,
         sScB, sScH, sTok, DKc, sTok, DKc, Hc, Bc * Hc, 0, 0);

    gemm(stream, AT, wo, T0, nullptr, MS, Dc, HDK, HDK, Dc, Dc, 0,0,0,0,0,0, 1, 1, 0, 0);
    add_ln_kernel<<<MS, 256, 0, stream>>>(X, T0, enc_ln1g + (long)l * Dc, enc_ln1b + (long)l * Dc, Dc);

    gemm(stream, X,  w1, FF, enc_b1 + (long)l * Fc, MS, Fc, Dc, Dc, Fc, Fc, 0,0,0,0,0,0, 1, 1, 0, 1);
    gemm(stream, FF, w2, T0, enc_b2 + (long)l * Dc, MS, Dc, Fc, Fc, Dc, Dc, 0,0,0,0,0,0, 1, 1, 0, 0);
    add_ln_kernel<<<MS, 256, 0, stream>>>(X, T0, enc_ln2g + (long)l * Dc, enc_ln2b + (long)l * Dc, Dc);
  }

  // =============================== decoder ===============================
  embed_kernel<<<MS, 256, 0, stream>>>(trg, emb_trg, Y, Tc, Dc, emb_scale);

  for (int l = 0; l < Nl; ++l) {
    const float* wq = dec_wq + (long)l * Dc * HDK;
    const float* wk = dec_wk + (long)l * Dc * HDK;
    const float* wv = dec_wv + (long)l * Dc * HDK;
    const float* wo = dec_wo + (long)l * HDK * Dc;
    const float* cq = dec_cq + (long)l * Dc * HDK;
    const float* ck = dec_ck + (long)l * Dc * HDK;
    const float* cv = dec_cv + (long)l * Dc * HDK;
    const float* co = dec_co + (long)l * HDK * Dc;
    const float* w1 = dec_w1 + (long)l * Dc * Fc;
    const float* w2 = dec_w2 + (long)l * Fc * Dc;

    // ---- masked self-attention (causal + trg pad) ----
    gemm(stream, Y, wq, Q,  nullptr, MS, HDK, Dc, Dc, HDK, HDK, 0,0,0,0,0,0, 1, 1, 0, 0);
    gemm(stream, Y, wk, Kb, nullptr, MS, HDK, Dc, Dc, HDK, HDK, 0,0,0,0,0,0, 1, 1, 0, 0);
    gemm(stream, Y, wv, Vb, nullptr, MS, HDK, Dc, Dc, HDK, HDK, 0,0,0,0,0,0, 1, 1, 0, 0);
    gemm(stream, Q, Kb, SC, nullptr, Tc, Tc, DKc, HDK, HDK, Tc,
         sTok, DKc, sTok, DKc, sScB, sScH, Hc, Bc * Hc, 1, 0);
    softmax_mask_kernel<<<dim3(Tc, Bc * Hc), 256, 0, stream>>>(
        SC, trg, Hc, Tc, Tc, 1, att_scale);
    gemm(stream, SC, Vb, AT, nullptr, Tc, DKc, Tc, Tc, HDK, HDK,
         sScB, sScH, sTok, DKc, sTok, DKc, Hc, Bc * Hc, 0, 0);
    gemm(stream, AT, wo, T0, nullptr, MS, Dc, HDK, HDK, Dc, Dc, 0,0,0,0,0,0, 1, 1, 0, 0);
    add_ln_kernel<<<MS, 256, 0, stream>>>(Y, T0, dec_ln1g + (long)l * Dc, dec_ln1b + (long)l * Dc, Dc);

    // ---- cross-attention (K/V from encoder output X, src pad mask) ----
    gemm(stream, Y, cq, Q,  nullptr, MS, HDK, Dc, Dc, HDK, HDK, 0,0,0,0,0,0, 1, 1, 0, 0);
    gemm(stream, X, ck, Kb, nullptr, MS, HDK, Dc, Dc, HDK, HDK, 0,0,0,0,0,0, 1, 1, 0, 0);
    gemm(stream, X, cv, Vb, nullptr, MS, HDK, Dc, Dc, HDK, HDK, 0,0,0,0,0,0, 1, 1, 0, 0);
    gemm(stream, Q, Kb, SC, nullptr, Tc, Sc, DKc, HDK, HDK, Sc,
         sTok, DKc, sTok, DKc, sScB, sScH, Hc, Bc * Hc, 1, 0);
    softmax_mask_kernel<<<dim3(Tc, Bc * Hc), 256, 0, stream>>>(
        SC, src, Hc, Tc, Sc, 0, att_scale);
    gemm(stream, SC, Vb, AT, nullptr, Tc, DKc, Sc, Sc, HDK, HDK,
         sScB, sScH, sTok, DKc, sTok, DKc, Hc, Bc * Hc, 0, 0);
    gemm(stream, AT, co, T0, nullptr, MS, Dc, HDK, HDK, Dc, Dc, 0,0,0,0,0,0, 1, 1, 0, 0);
    add_ln_kernel<<<MS, 256, 0, stream>>>(Y, T0, dec_ln2g + (long)l * Dc, dec_ln2b + (long)l * Dc, Dc);

    // ---- FFN ----
    gemm(stream, Y,  w1, FF, dec_b1 + (long)l * Fc, MS, Fc, Dc, Dc, Fc, Fc, 0,0,0,0,0,0, 1, 1, 0, 1);
    gemm(stream, FF, w2, T0, dec_b2 + (long)l * Dc, MS, Dc, Fc, Fc, Dc, Dc, 0,0,0,0,0,0, 1, 1, 0, 0);
    add_ln_kernel<<<MS, 256, 0, stream>>>(Y, T0, dec_ln3g + (long)l * Dc, dec_ln3b + (long)l * Dc, Dc);
  }

  // ---- final projection: [MS,512] x [512,32000] -> d_out ----
  gemm(stream, Y, proj, out, nullptr, MS, Vc, Dc, Dc, Vc, Vc, 0,0,0,0,0,0, 1, 1, 0, 0);
}
